// SelfAttention_72997264163219
// MI455X (gfx1250) — matrix operand order repaired and checked by reading, not compiled, not run on hardware
//
#include <hip/hip_runtime.h>
#include <hip/hip_bf16.h>

typedef __attribute__((ext_vector_type(16))) __bf16 v16bf;
typedef __attribute__((ext_vector_type(8)))  __bf16 v8bf;
typedef __attribute__((ext_vector_type(8)))  float  v8f;
typedef __attribute__((ext_vector_type(4)))  unsigned int u32x4;
typedef __attribute__((ext_vector_type(8)))  int i32x8;
typedef __attribute__((ext_vector_type(4)))  int i32x4;

#define FDIM 1024
#define NSEQ 2048
#define NBATCH 8
#define MTOT (NBATCH * NSEQ)   // 16384
#define APAD 8
#define BK 64
#define NSTAGE (FDIM / BK)     // 16

__device__ __forceinline__ __bf16 f2bf(float f) {
    unsigned int u = __builtin_bit_cast(unsigned int, f);
    unsigned int r = u + 0x7FFFu + ((u >> 16) & 1u);   // round to nearest even
    unsigned short h = (unsigned short)(r >> 16);
    return __builtin_bit_cast(__bf16, h);
}

#define ASYNC_WAIT(n) asm volatile("s_wait_asynccnt " n ::: "memory")

__device__ __forceinline__ unsigned lds_addr(const void* p) {
    return (unsigned)(uintptr_t)p;   // low 32 bits of shared aperture = LDS offset
}

// ---- CDNA5 Tensor Data Mover: one-instruction 2D tile DMA global->LDS ------
// Tile: 128 rows x 128B (BK=64 bf16), global row stride 2048B, LDS pad 16B
// per 128B row  ->  [128][72] bf16 layout (144B = 36-dword stride, walks all
// 64 LDS banks over the 16 fragment rows).
__device__ __forceinline__ void tdm_load_tile(const void* gsrc, unsigned ldsdst) {
    const unsigned long long ga = (unsigned long long)(uintptr_t)gsrc;
    u32x4 g0;
    g0[0] = 1u;                                   // count=1, user descriptor
    g0[1] = ldsdst;                               // lds_addr
    g0[2] = (unsigned)(ga & 0xFFFFFFFFu);         // global_addr[31:0]
    g0[3] = (unsigned)((ga >> 32) & 0x1FFFFFFu)   // global_addr[56:32]
            | 0x80000000u;                        // type=2 ("image")
    i32x8 g1;
    g1[0] = (int)((2u << 16)    // data_size = 4B
                | (1u << 20)    // pad_enable
                | (4u << 22)    // pad_interval: 32 dwords (128B)
                | (3u << 25));  // pad_amount:   4 dwords (16B)
    g1[1] = (int)(512u << 16);  // tensor_dim0 = 512 dwords (2048B row)
    g1[2] = (int)(128u << 16);  // tensor_dim1 = 128 rows
    g1[3] = (int)(32u << 16);   // tile_dim0   = 32 dwords (128B)
    g1[4] = 128;                // tile_dim1   = 128 rows, tile_dim2 = 0
    g1[5] = 512;                // tensor_dim0_stride = 512 dwords
    g1[6] = 0;
    g1[7] = 0;
    const i32x4 gz4 = {0, 0, 0, 0};
    const i32x8 gz8 = {0, 0, 0, 0, 0, 0, 0, 0};
    __builtin_amdgcn_tensor_load_to_lds(g0, g1, gz4, gz4, gz8, 0);
}

// ---------------------------------------------------------------------------
// Kernel 0: fp32 -> bf16 conversion pass.  n8 = element_count / 8.
// ---------------------------------------------------------------------------
__global__ __launch_bounds__(256) void cvt_bf16_kernel(
    const float* __restrict__ src, __bf16* __restrict__ dst, int n8)
{
    const int i = blockIdx.x * 256 + threadIdx.x;
    if (i >= n8) return;
    const float4* s4 = (const float4*)src + (size_t)i * 2;
    const float4 a = s4[0];
    const float4 b = s4[1];
    union { v8bf v; __bf16 e[8]; } o;
    o.e[0] = f2bf(a.x); o.e[1] = f2bf(a.y); o.e[2] = f2bf(a.z); o.e[3] = f2bf(a.w);
    o.e[4] = f2bf(b.x); o.e[5] = f2bf(b.y); o.e[6] = f2bf(b.z); o.e[7] = f2bf(b.w);
    *(v8bf*)(dst + (size_t)i * 8) = o.v;
}

// ---------------------------------------------------------------------------
// Kernel 1: fused QKV projection.  TDM double-buffered tile staging.
// blockIdx.z: 0 = Q (pre-scaled by 1/sqrt(D)), 1 = K, 2 = V (stored transposed)
// 128x128 tile / block, 8 waves, BK=64 (two WMMA K-steps per stage).
// Epilogue bounces the result tile through LDS for coalesced b128 stores.
// ---------------------------------------------------------------------------
__global__ __launch_bounds__(256) void qkv_proj_kernel(
    const __bf16* __restrict__ xb,
    const __bf16* __restrict__ Wqb, const float* __restrict__ bq,
    const __bf16* __restrict__ Wkb, const float* __restrict__ bk,
    const __bf16* __restrict__ Wvb, const float* __restrict__ bv,
    __bf16* __restrict__ qout, __bf16* __restrict__ kout,
    __bf16* __restrict__ vtout)
{
    __shared__ union {
        __bf16 tiles[2][2][128][BK + APAD];   // [mat A/B][buf][row][col]
        __bf16 ct[128][128 + APAD];           // epilogue staging
    } sm;

    const int z = blockIdx.z;
    const __bf16* Wb   = (z == 0) ? Wqb : (z == 1) ? Wkb : Wvb;
    const float*  bias = (z == 0) ? bq  : (z == 1) ? bk  : bv;

    const int tid  = threadIdx.x;
    const int lane = tid & 31;
    const int wave = tid >> 5;
    const int wm   = wave >> 2;      // 0..1 -> M offset wm*64
    const int wn   = wave & 3;       // 0..3 -> N offset wn*32
    const int cl   = lane & 15;
    const int hf   = lane >> 4;

    const int m0 = blockIdx.y * 128;
    const int n0 = blockIdx.x * 128;

    const __bf16* A0 = xb + (size_t)m0 * FDIM;
    const __bf16* B0 = Wb + (size_t)n0 * FDIM;

    // wave 0 DMAs the A tile, wave 1 the B tile (EXEC-independent TDM issue)
    auto issue_stage = [&](int s) {
        const int buf = s & 1;
        if (wave == 0)
            tdm_load_tile(A0 + s * BK, lds_addr(&sm.tiles[0][buf][0][0]));
        else
            tdm_load_tile(B0 + s * BK, lds_addr(&sm.tiles[1][buf][0][0]));
    };

    v8f acc[4][2] = {};
    if (wave < 2) issue_stage(0);

    for (int s = 0; s < NSTAGE; ++s) {
        if (wave < 2) {
            if (s + 1 < NSTAGE) {
                issue_stage(s + 1);
                __builtin_amdgcn_s_wait_tensorcnt(1);  // stage s landed
            } else {
                __builtin_amdgcn_s_wait_tensorcnt(0);
            }
        }
        __syncthreads();

        const int buf = s & 1;
        #pragma unroll
        for (int kk = 0; kk < BK; kk += 32) {
            v16bf afr[4], bfr[2];
            const int kbA = kk + (hf << 4);   // +0 / +16
            #pragma unroll
            for (int i = 0; i < 4; ++i) {
                const int r = wm * 64 + i * 16 + cl;
                union { v16bf v; v8bf h[2]; } u;
                u.h[0] = *(const v8bf*)&sm.tiles[0][buf][r][kbA];
                u.h[1] = *(const v8bf*)&sm.tiles[0][buf][r][kbA + 8];
                afr[i] = u.v;
            }
            const int kbB = kk + (hf << 4);   // +0 / +16
            #pragma unroll
            for (int j = 0; j < 2; ++j) {
                const int c = wn * 32 + j * 16 + cl;
                union { v16bf v; v8bf h[2]; } u;
                u.h[0] = *(const v8bf*)&sm.tiles[1][buf][c][kbB];
                u.h[1] = *(const v8bf*)&sm.tiles[1][buf][c][kbB + 8];
                bfr[j] = u.v;
            }
            #pragma unroll
            for (int i = 0; i < 4; ++i)
                #pragma unroll
                for (int j = 0; j < 2; ++j)
                    acc[i][j] = __builtin_amdgcn_wmma_f32_16x16x32_bf16(
                        false, afr[i], false, bfr[j], (short)0, acc[i][j],
                        false, false);
        }
        __syncthreads();
    }

    // ---- epilogue: bias (+Q scale), stage tile in LDS, coalesced stores ----
    const float qscale = (z == 0) ? 0.03125f : 1.0f;  // 1/sqrt(1024)
    #pragma unroll
    for (int j = 0; j < 2; ++j) {
        const int cg = n0 + wn * 32 + j * 16 + cl;
        const int cl_loc = wn * 32 + j * 16 + cl;
        const float bval = bias[cg];
        #pragma unroll
        for (int i = 0; i < 4; ++i) {
            #pragma unroll
            for (int r = 0; r < 8; ++r) {
                const int rl = wm * 64 + i * 16 + r + (hf << 3);
                const __bf16 h = f2bf((acc[i][j][r] + bval) * qscale);
                if (z < 2) sm.ct[rl][cl_loc] = h;       // [token][feature]
                else       sm.ct[cl_loc][rl] = h;       // [feature][token]
            }
        }
    }
    __syncthreads();

    const int trow = tid >> 1;            // 0..127
    const int c0   = (tid & 1) << 6;      // 0 / 64
    if (z < 2) {
        __bf16* dst = ((z == 0) ? qout : kout)
                      + (size_t)(m0 + trow) * FDIM + n0 + c0;
        #pragma unroll
        for (int i = 0; i < 64; i += 8)
            *(v8bf*)(dst + i) = *(const v8bf*)&sm.ct[trow][c0 + i];
    } else {
        const int bidx = m0 >> 11;            // batch
        const int tok0 = (m0 & (NSEQ - 1)) + c0;
        __bf16* dst = vtout + ((size_t)bidx * FDIM + n0 + trow) * NSEQ + tok0;
        #pragma unroll
        for (int i = 0; i < 64; i += 8)
            *(v8bf*)(dst + i) = *(const v8bf*)&sm.ct[trow][c0 + i];
    }
}

// ---------------------------------------------------------------------------
// Kernel 2: flash attention.  Block = 16 query rows of one batch, 8 waves.
// Wave w owns keys [kb0+16w, +16) for score WMMAs and features [128w, +128)
// for the PV WMMAs.  Q staged via async copies; next K block prefetched.
// Score phase uses 4 independent WMMA accumulator chains for pipelining.
// ---------------------------------------------------------------------------
__global__ __launch_bounds__(256) void attn_kernel(
    const __bf16* __restrict__ qb, const __bf16* __restrict__ kbuf,
    const __bf16* __restrict__ vt, float* __restrict__ out)
{
    __shared__ __bf16 Qs[16][FDIM + APAD];
    __shared__ __bf16 Ps[16][128 + APAD];
    __shared__ float wmax[8][16];
    __shared__ float wsum[8][16];
    __shared__ float mrun[16], lrun[16], alpha[16];

    const int tid  = threadIdx.x;
    const int lane = tid & 31;
    const int wave = tid >> 5;
    const int cl   = lane & 15;
    const int hf   = lane >> 4;
    const int b    = blockIdx.y;
    const int q0   = blockIdx.x * 16;

    // stage the 16 x 1024 bf16 Q tile into LDS with async copies
    {
        const int row = tid >> 4;
        const int c0  = (tid & 15) << 6;             // 64 elems = 128 bytes
        const __bf16* src = qb + (size_t)(b * NSEQ + q0 + row) * FDIM + c0;
        const unsigned dst = lds_addr(&Qs[row][c0]);
        // INST_OFFSET is added to both the LDS and global address (ISA 10.x)
        asm volatile(
            "global_load_async_to_lds_b128 %0, %1, off\n\t"
            "global_load_async_to_lds_b128 %0, %1, off offset:16\n\t"
            "global_load_async_to_lds_b128 %0, %1, off offset:32\n\t"
            "global_load_async_to_lds_b128 %0, %1, off offset:48\n\t"
            "global_load_async_to_lds_b128 %0, %1, off offset:64\n\t"
            "global_load_async_to_lds_b128 %0, %1, off offset:80\n\t"
            "global_load_async_to_lds_b128 %0, %1, off offset:96\n\t"
            "global_load_async_to_lds_b128 %0, %1, off offset:112"
            :: "v"(dst), "v"(src) : "memory");
        ASYNC_WAIT("0x0");
    }
    if (tid < 16) { mrun[tid] = -1e30f; lrun[tid] = 0.0f; }

    v8f o[8] = {};
    __syncthreads();

    for (int kb0 = 0; kb0 < NSEQ; kb0 += 128) {
        const __bf16* krow =
            kbuf + (size_t)(b * NSEQ + kb0 + wave * 16 + cl) * FDIM + hf * 16;
        if (kb0 + 128 < NSEQ) {   // prefetch next key block (-> global_prefetch)
            __builtin_prefetch(krow + (size_t)128 * FDIM, 0, 1);
            __builtin_prefetch(krow + (size_t)128 * FDIM + 512, 0, 1);
        }

        // ---- scores: 4 independent WMMA chains over the D=1024 reduction ----
        v8f s0 = {}, s1 = {}, s2 = {}, s3 = {};
        #pragma unroll 2
        for (int kk = 0; kk < FDIM; kk += 128) {
            #pragma unroll
            for (int c = 0; c < 4; ++c) {
                const int k = kk + c * 32;
                union { v16bf v; v8bf h[2]; } ua, ub;
                ua.h[0] = *(const v8bf*)&Qs[cl][k + hf * 16];
                ua.h[1] = *(const v8bf*)&Qs[cl][k + hf * 16 + 8];
                ub.h[0] = *(const v8bf*)(krow + k);
                ub.h[1] = *(const v8bf*)(krow + k + 8);
                v8f& sc = (c == 0) ? s0 : (c == 1) ? s1 : (c == 2) ? s2 : s3;
                sc = __builtin_amdgcn_wmma_f32_16x16x32_bf16(
                    false, ua.v, false, ub.v, (short)0, sc, false, false);
            }
        }
        const v8f s = (s0 + s1) + (s2 + s3);

        // ---- per-wave row max (cross-lane within each 16-lane half) ----
        #pragma unroll
        for (int r = 0; r < 8; ++r) {
            float m = s[r];
            m = fmaxf(m, __shfl_xor(m, 1, 32));
            m = fmaxf(m, __shfl_xor(m, 2, 32));
            m = fmaxf(m, __shfl_xor(m, 4, 32));
            m = fmaxf(m, __shfl_xor(m, 8, 32));
            if (cl == 0) wmax[wave][r + (hf << 3)] = m;
        }
        __syncthreads();

        if (tid < 16) {
            float bm = wmax[0][tid];
            #pragma unroll
            for (int w = 1; w < 8; ++w) bm = fmaxf(bm, wmax[w][tid]);
            const float mo = mrun[tid];
            const float mn = fmaxf(mo, bm);
            alpha[tid] = __expf(mo - mn);
            mrun[tid]  = mn;
        }
        __syncthreads();

        // ---- exponentiate, write P tile to LDS, partial row sums ----
        #pragma unroll
        for (int r = 0; r < 8; ++r) {
            const int row = r + (hf << 3);
            const float p = __expf(s[r] - mrun[row]);
            Ps[row][wave * 16 + cl] = f2bf(p);
            float sum = p;
            sum += __shfl_xor(sum, 1, 32);
            sum += __shfl_xor(sum, 2, 32);
            sum += __shfl_xor(sum, 4, 32);
            sum += __shfl_xor(sum, 8, 32);
            if (cl == 0) wsum[wave][row] = sum;
        }
        // rescale running O by alpha
        #pragma unroll
        for (int j = 0; j < 8; ++j)
            #pragma unroll
            for (int r = 0; r < 8; ++r)
                o[j][r] *= alpha[r + (hf << 3)];
        __syncthreads();

        if (tid < 16) {
            float sa = 0.0f;
            #pragma unroll
            for (int w = 0; w < 8; ++w) sa += wsum[w][tid];
            lrun[tid] = lrun[tid] * alpha[tid] + sa;
        }

        // ---- O += P(16x128) . V_block(128 x 128-feature slice) ----
        const __bf16* vbase = vt + (size_t)b * FDIM * NSEQ;
        #pragma unroll
        for (int kk2 = 0; kk2 < 4; ++kk2) {
            union { v16bf v; v8bf h[2]; } ua;
            ua.h[0] = *(const v8bf*)&Ps[cl][kk2 * 32 + hf * 16];
            ua.h[1] = *(const v8bf*)&Ps[cl][kk2 * 32 + hf * 16 + 8];
            #pragma unroll
            for (int j = 0; j < 8; ++j) {
                const int feat = wave * 128 + j * 16 + cl;
                const __bf16* vrow =
                    vbase + (size_t)feat * NSEQ + kb0 + kk2 * 32 + hf * 16;
                union { v16bf v; v8bf h[2]; } ub;
                ub.h[0] = *(const v8bf*)(vrow);
                ub.h[1] = *(const v8bf*)(vrow + 8);
                o[j] = __builtin_amdgcn_wmma_f32_16x16x32_bf16(
                    false, ua.v, false, ub.v, (short)0, o[j], false, false);
            }
        }
        __syncthreads();
    }

    // ---- normalize by row sums and store fp32 output ----
    #pragma unroll
    for (int j = 0; j < 8; ++j) {
        const int feat = wave * 128 + j * 16 + cl;
        #pragma unroll
        for (int r = 0; r < 8; ++r) {
            const int row = r + (hf << 3);
            out[(size_t)(b * NSEQ + q0 + row) * FDIM + feat] =
                o[j][r] / lrun[row];
        }
    }
}

extern "C" void kernel_launch(void* const* d_in, const int* in_sizes, int n_in,
                              void* d_out, int out_size, void* d_ws, size_t ws_size,
                              hipStream_t stream) {
    const float* x  = (const float*)d_in[0];
    const float* Wq = (const float*)d_in[1];
    const float* bq = (const float*)d_in[2];
    const float* Wk = (const float*)d_in[3];
    const float* bk = (const float*)d_in[4];
    const float* Wv = (const float*)d_in[5];
    const float* bv = (const float*)d_in[6];
    float* out = (float*)d_out;

    __bf16* xb   = (__bf16*)d_ws;                        // 32 MB
    __bf16* Wqb  = xb   + (size_t)MTOT * FDIM;           //  2 MB
    __bf16* Wkb  = Wqb  + (size_t)FDIM * FDIM;
    __bf16* Wvb  = Wkb  + (size_t)FDIM * FDIM;
    __bf16* qb   = Wvb  + (size_t)FDIM * FDIM;           // 32 MB
    __bf16* kbuf = qb   + (size_t)MTOT * FDIM;           // 32 MB
    __bf16* vt   = kbuf + (size_t)MTOT * FDIM;           // 32 MB (transposed)

    const int n8x = (MTOT * FDIM) / 8;       // 2M vec8 chunks
    const int n8w = (FDIM * FDIM) / 8;       // 128K vec8 chunks
    cvt_bf16_kernel<<<dim3((n8x + 255) / 256), dim3(256), 0, stream>>>(x,  xb,  n8x);
    cvt_bf16_kernel<<<dim3((n8w + 255) / 256), dim3(256), 0, stream>>>(Wq, Wqb, n8w);
    cvt_bf16_kernel<<<dim3((n8w + 255) / 256), dim3(256), 0, stream>>>(Wk, Wkb, n8w);
    cvt_bf16_kernel<<<dim3((n8w + 255) / 256), dim3(256), 0, stream>>>(Wv, Wvb, n8w);

    dim3 gridP(FDIM / 128, MTOT / 128, 3);
    qkv_proj_kernel<<<gridP, dim3(256), 0, stream>>>(
        xb, Wqb, bq, Wkb, bk, Wvb, bv, qb, kbuf, vt);

    dim3 gridA(NSEQ / 16, NBATCH);
    attn_kernel<<<gridA, dim3(256), 0, stream>>>(qb, kbuf, vt, out);
}
